// MyRnn_43413529428167
// MI455X (gfx1250) — matrix operand order repaired and checked by reading, not compiled, not run on hardware
//
#include <hip/hip_runtime.h>

#define H_   128
#define T_   28
#define GATE 512   // 4*H

typedef __attribute__((ext_vector_type(16))) __bf16 v16bf;
typedef __attribute__((ext_vector_type(8)))  __bf16 v8bf;
typedef __attribute__((ext_vector_type(8)))  float  v8f;

__device__ __forceinline__ v16bf cat16(v8bf lo, v8bf hi) {
  return __builtin_shufflevector(lo, hi, 0,1,2,3,4,5,6,7,8,9,10,11,12,13,14,15);
}

// A fragment (16x32 bf16): lane<16 -> row=lane, K 0..15;
// lane>=16 -> row=lane-16, K 16..31. Source row-major, stride in halves.
__device__ __forceinline__ v16bf a_frag(const __bf16* base, int stride, int kbase, int lane) {
  int m  = lane & 15;
  int kb = kbase + ((lane >> 4) << 4);
  const __bf16* p = base + m * stride + kb;
  v8bf lo = *(const v8bf*)p;          // K kb..kb+7
  v8bf hi = *(const v8bf*)(p + 8);    // K kb+8..kb+15
  return cat16(lo, hi);
}

// B fragment (32x16 bf16): B(k,n)=W[ncol0+n][kbase+k], W row-major [512][K].
// lane<16: N=lane, K kbase..kbase+15 ; lane>=16: N=lane-16, K kbase+16..kbase+31
// -> 16 contiguous halves per lane = one 32B read.
__device__ __forceinline__ v16bf b_frag(const __bf16* W, int K, int ncol0, int kbase, int lane) {
  int n  = ncol0 + (lane & 15);
  int kb = kbase + ((lane >> 4) << 4);
  return *(const v16bf*)(W + (size_t)n * K + kb);
}

__device__ __forceinline__ float sigm(float x)      { return 1.0f / (1.0f + __expf(-x)); }
__device__ __forceinline__ float tanh_fast(float x) { return 1.0f - 2.0f / (__expf(2.0f * x) + 1.0f); }

// f32 -> bf16 weight conversion with optional K padding (28 -> 32 for layer0 w_ih)
__global__ void convert_pad(const float* __restrict__ src, __bf16* __restrict__ dst,
                            int rows, int sk, int dk) {
  int i = blockIdx.x * blockDim.x + threadIdx.x;
  if (i >= rows * dk) return;
  int r = i / dk, k = i % dk;
  dst[i] = (k < sk) ? (__bf16)src[r * sk + k] : (__bf16)0.0f;
}

// Persistent bidirectional-capable LSTM direction kernel.
// Block = 16 batch rows, 256 threads (8 waves). Wave w owns gate columns
// [64w, 64w+64) for the GEMM and h columns [16w, 16w+16) for the cell math.
// w_hh is held in VGPRs as 16 WMMA B-fragments for the whole recurrence.
template<int LAYER>
__global__ void __launch_bounds__(256)
lstm_dir(const float*  __restrict__ x0,    // LAYER==0: [N][784] f32
         const __bf16* __restrict__ xin,   // LAYER==1: y0 [N][T][256] bf16
         const __bf16* __restrict__ wih,   // [512][KIN] bf16 (KIN=32 padded / 256)
         const __bf16* __restrict__ whh,   // [512][128] bf16
         const float*  __restrict__ bih,
         const float*  __restrict__ bhh,
         __bf16* __restrict__ yout,        // LAYER==0: y0 [N][T][256]
         float*  __restrict__ ylast,       // LAYER==1: [N][256]
         int reverse, int nsteps, int dirOff)
{
  constexpr int HSTR = 136;                          // padded h row stride (halves)
  constexpr int GSTR = 516;                          // padded gates row stride (f32)
  constexpr int XSTR = (LAYER == 0) ? 904 : 264;     // padded x row stride (halves)

  const int tid  = threadIdx.x;
  const int lane = tid & 31;
  const int wave = tid >> 5;
  const int row0 = blockIdx.x * 16;

  __shared__ alignas(32) __bf16 sh_h[16 * HSTR];
  __shared__ alignas(32) float  sh_g[16 * GSTR];
  __shared__ alignas(32) __bf16 sh_x[16 * XSTR];

  // zero initial hidden state
  for (int i = tid; i < 16 * HSTR; i += 256) sh_h[i] = (__bf16)0.0f;

  if constexpr (LAYER == 0) {
    // stage the full input sequence for this row tile, K zero-padded 28 -> 32
    for (int i = tid; i < 16 * T_ * 32; i += 256) {
      int r = i / (T_ * 32), rem = i % (T_ * 32);
      int t = rem >> 5, k = rem & 31;
      float v = (k < 28) ? x0[(size_t)(row0 + r) * 784 + t * 28 + k] : 0.0f;
      sh_x[r * XSTR + rem] = (__bf16)v;
    }
  }

  // hold w_hh in registers: 4 gate tiles x 4 K-chunks = 16 fragments (128 VGPRs)
  const int gbase = wave * 64;
  v16bf Bh[4][4];
#pragma unroll
  for (int i = 0; i < 4; ++i)
#pragma unroll
    for (int kc = 0; kc < 4; ++kc)
      Bh[i][kc] = b_frag(whh, H_, gbase + i * 16, kc * 32, lane);

  v16bf Bi0[4];
  if constexpr (LAYER == 0) {
#pragma unroll
    for (int i = 0; i < 4; ++i) Bi0[i] = b_frag(wih, 32, gbase + i * 16, 0, lane);
  }

  // per-lane cell constants: this lane's h column and fused biases
  const int ecol = wave * 16 + (lane & 15);
  const int ero  = lane >> 4;
  const float bi = bih[ecol        ] + bhh[ecol        ];
  const float bf = bih[ecol + 128  ] + bhh[ecol + 128  ];
  const float bg = bih[ecol + 2*128] + bhh[ecol + 2*128];
  const float bo = bih[ecol + 3*128] + bhh[ecol + 3*128];

  float c[8];
#pragma unroll
  for (int i = 0; i < 8; ++i) c[i] = 0.0f;

  for (int t = 0; t < nsteps; ++t) {
    const int tt = reverse ? (T_ - 1 - t) : t;
    __syncthreads();   // prev cell phase done: sh_h valid, sh_g/sh_x reusable

    if constexpr (LAYER == 1) {
      // stage x_t = y0[:, tt, :] (16 x 256 bf16) cooperatively
      int r = tid >> 4, c0 = (tid & 15) << 4;
      const __bf16* src = xin + ((size_t)(row0 + r) * T_ + tt) * 256 + c0;
      __bf16* dst = sh_x + r * XSTR + c0;
      *(v8bf*)dst       = *(const v8bf*)src;
      *(v8bf*)(dst + 8) = *(const v8bf*)(src + 8);
      if (t + 1 < nsteps) {  // prefetch next timestep tile into caches
        int nt = reverse ? (tt - 1) : (tt + 1);
        __builtin_prefetch(xin + ((size_t)(row0 + r) * T_ + nt) * 256 + c0, 0, 1);
      }
      __syncthreads();
    }

    v8f acc[4];
    const v8f vz = {0.f, 0.f, 0.f, 0.f, 0.f, 0.f, 0.f, 0.f};
#pragma unroll
    for (int i = 0; i < 4; ++i) acc[i] = vz;

    // input contribution: x_t @ w_ih^T
    if constexpr (LAYER == 0) {
      v16bf a = a_frag(sh_x, XSTR, tt * 32, lane);
#pragma unroll
      for (int i = 0; i < 4; ++i)
        acc[i] = __builtin_amdgcn_wmma_f32_16x16x32_bf16(
            false, a, false, Bi0[i], (short)0, acc[i], false, false);
    } else {
      // hoist all 8 A-fragments so the ds_loads issue back-to-back and the
      // 32-WMMA chain runs behind a single dscnt wait
      v16bf ax[8];
#pragma unroll
      for (int kc = 0; kc < 8; ++kc) ax[kc] = a_frag(sh_x, XSTR, kc * 32, lane);
#pragma unroll
      for (int kc = 0; kc < 8; ++kc) {
#pragma unroll
        for (int i = 0; i < 4; ++i) {
          v16bf b = b_frag(wih, 256, gbase + i * 16, kc * 32, lane);
          acc[i] = __builtin_amdgcn_wmma_f32_16x16x32_bf16(
              false, ax[kc], false, b, (short)0, acc[i], false, false);
        }
      }
    }

    // recurrent contribution: h_{t-1} @ w_hh^T (weights resident in VGPRs);
    // hoist the 4 A-fragments ahead of the 16-WMMA chain
    {
      v16bf ah[4];
#pragma unroll
      for (int kc = 0; kc < 4; ++kc) ah[kc] = a_frag(sh_h, HSTR, kc * 32, lane);
#pragma unroll
      for (int kc = 0; kc < 4; ++kc)
#pragma unroll
        for (int i = 0; i < 4; ++i)
          acc[i] = __builtin_amdgcn_wmma_f32_16x16x32_bf16(
              false, ah[kc], false, Bh[i][kc], (short)0, acc[i], false, false);
    }

    // scatter gate accumulators to LDS (C/D layout: lane<16 N=lane M=j, else M=j+8)
#pragma unroll
    for (int i = 0; i < 4; ++i) {
      int n = gbase + i * 16 + (lane & 15);
#pragma unroll
      for (int j = 0; j < 8; ++j) {
        int m = j + ((lane >> 4) << 3);
        sh_g[m * GSTR + n] = acc[i][j];
      }
    }
    __syncthreads();

    // LSTM cell (PyTorch order i,f,g,o) on this wave's 16 h columns
#pragma unroll
    for (int rr = 0; rr < 8; ++rr) {
      int r = rr * 2 + ero;
      float ig = sh_g[r * GSTR + ecol          ] + bi;
      float fg = sh_g[r * GSTR + ecol + 128    ] + bf;
      float gg = sh_g[r * GSTR + ecol + 2 * 128] + bg;
      float og = sh_g[r * GSTR + ecol + 3 * 128] + bo;
      float cn = sigm(fg) * c[rr] + sigm(ig) * tanh_fast(gg);
      c[rr] = cn;
      float h = sigm(og) * tanh_fast(cn);
      sh_h[r * HSTR + ecol] = (__bf16)h;
      if constexpr (LAYER == 0) {
        yout[((size_t)(row0 + r) * T_ + tt) * 256 + dirOff + ecol] = (__bf16)h;
      } else {
        if (t == nsteps - 1)
          ylast[(size_t)(row0 + r) * 256 + dirOff + ecol] = h;
      }
    }
  }
}

// out[n][j] = b_out[j] + ylast[n] . w_out[j]   (10 classes; trivial cost)
__global__ void classifier(const float* __restrict__ ylast, const float* __restrict__ wout,
                           const float* __restrict__ bout, float* __restrict__ out, int n) {
  int idx = blockIdx.x * blockDim.x + threadIdx.x;
  if (idx >= n * 10) return;
  int r = idx / 10, j = idx % 10;
  const float* yr = ylast + (size_t)r * 256;
  const float* wr = wout + j * 256;
  float s = bout[j];
#pragma unroll 4
  for (int k = 0; k < 256; ++k) s += yr[k] * wr[k];
  out[idx] = s;
}

extern "C" void kernel_launch(void* const* d_in, const int* in_sizes, int n_in,
                              void* d_out, int out_size, void* d_ws, size_t ws_size,
                              hipStream_t stream) {
  (void)n_in; (void)out_size; (void)ws_size;
  const float* x       = (const float*)d_in[0];
  const float* wih_l0  = (const float*)d_in[1];
  const float* whh_l0  = (const float*)d_in[2];
  const float* bih_l0  = (const float*)d_in[3];
  const float* bhh_l0  = (const float*)d_in[4];
  const float* wih_l0r = (const float*)d_in[5];
  const float* whh_l0r = (const float*)d_in[6];
  const float* bih_l0r = (const float*)d_in[7];
  const float* bhh_l0r = (const float*)d_in[8];
  const float* wih_l1  = (const float*)d_in[9];
  const float* whh_l1  = (const float*)d_in[10];
  const float* bih_l1  = (const float*)d_in[11];
  const float* bhh_l1  = (const float*)d_in[12];
  const float* wih_l1r = (const float*)d_in[13];
  const float* whh_l1r = (const float*)d_in[14];
  const float* bih_l1r = (const float*)d_in[15];
  const float* bhh_l1r = (const float*)d_in[16];
  const float* w_out   = (const float*)d_in[17];
  const float* b_out   = (const float*)d_in[18];

  const int N = in_sizes[0] / 784;   // 8192

  char* ws = (char*)d_ws;
  size_t off = 0;
  auto take = [&](size_t bytes) -> void* {
    void* p = ws + off;
    off = (off + bytes + 255) & ~(size_t)255;
    return p;
  };
  __bf16* y0     = (__bf16*)take((size_t)N * T_ * 256 * 2);  // bf16 bi-LSTM layer0 output
  float*  y1last = (float*) take((size_t)N * 256 * 4);       // f32 last-step layer1 output
  __bf16* cwih0f = (__bf16*)take(512 * 32 * 2);
  __bf16* cwih0b = (__bf16*)take(512 * 32 * 2);
  __bf16* cwhh0f = (__bf16*)take(512 * 128 * 2);
  __bf16* cwhh0b = (__bf16*)take(512 * 128 * 2);
  __bf16* cwih1f = (__bf16*)take(512 * 256 * 2);
  __bf16* cwih1b = (__bf16*)take(512 * 256 * 2);
  __bf16* cwhh1f = (__bf16*)take(512 * 128 * 2);
  __bf16* cwhh1b = (__bf16*)take(512 * 128 * 2);

  auto conv = [&](const float* s, __bf16* d, int rows, int sk, int dk) {
    int tot = rows * dk;
    convert_pad<<<(tot + 255) / 256, 256, 0, stream>>>(s, d, rows, sk, dk);
  };
  conv(wih_l0,  cwih0f, 512, 28, 32);
  conv(wih_l0r, cwih0b, 512, 28, 32);
  conv(whh_l0,  cwhh0f, 512, 128, 128);
  conv(whh_l0r, cwhh0b, 512, 128, 128);
  conv(wih_l1,  cwih1f, 512, 256, 256);
  conv(wih_l1r, cwih1b, 512, 256, 256);
  conv(whh_l1,  cwhh1f, 512, 128, 128);
  conv(whh_l1r, cwhh1b, 512, 128, 128);

  dim3 grid(N / 16), blk(256);
  // layer 0: full sequences, write y0 (forward cols 0..127, backward 128..255)
  lstm_dir<0><<<grid, blk, 0, stream>>>(x, nullptr, cwih0f, cwhh0f, bih_l0,  bhh_l0,
                                        y0, nullptr, 0, T_, 0);
  lstm_dir<0><<<grid, blk, 0, stream>>>(x, nullptr, cwih0b, cwhh0b, bih_l0r, bhh_l0r,
                                        y0, nullptr, 1, T_, 128);
  // layer 1: forward needs full recurrence but only last h; backward time-aligned
  // output at t=T-1 is exactly one reverse step from zero state.
  lstm_dir<1><<<grid, blk, 0, stream>>>(nullptr, y0, cwih1f, cwhh1f, bih_l1,  bhh_l1,
                                        nullptr, y1last, 0, T_, 0);
  lstm_dir<1><<<grid, blk, 0, stream>>>(nullptr, y0, cwih1b, cwhh1b, bih_l1r, bhh_l1r,
                                        nullptr, y1last, 1, 1, 128);

  classifier<<<(N * 10 + 255) / 256, 256, 0, stream>>>(y1last, w_out, b_out,
                                                       (float*)d_out, N);
}
